// AttentionBlock_16011638079856
// MI455X (gfx1250) — matrix operand order repaired and checked by reading, not compiled, not run on hardware
//
#include <hip/hip_runtime.h>
#include <hip/hip_bf16.h>
#include <math.h>

typedef __attribute__((ext_vector_type(16))) _Float16 v16h;
typedef __attribute__((ext_vector_type(8)))  _Float16 v8h;
typedef __attribute__((ext_vector_type(4)))  _Float16 v4h;
typedef __attribute__((ext_vector_type(8)))  float    v8f;

#define WMMA16x16x32(A, B, C) \
  __builtin_amdgcn_wmma_f32_16x16x32_f16(false, (A), false, (B), (short)0, (C), false, false)

// Problem constants: B=16, C=512, H=W=32 -> L=1024, 32 groups, 8 heads, hd=64.
#define CC   512
#define LL   1024
#define NH   8
#define HD   64

// ===========================================================================
// Kernel 1: GroupNorm (f32 in) -> normalized activations as f16, (b, c, l)
// ===========================================================================
__global__ __launch_bounds__(256)
void gn_kernel(const float* __restrict__ x, const float* __restrict__ gw,
               const float* __restrict__ gb, _Float16* __restrict__ h16) {
  const int g = blockIdx.x;            // 0..31
  const int b = blockIdx.y;            // 0..15
  const int t = threadIdx.x;           // 0..255
  const float* xp = x + ((size_t)b * CC + g * 16) * LL;   // 16384 floats

  float s = 0.f, ss = 0.f;
  for (int i = t; i < 4096; i += 256) {
    float4 v = ((const float4*)xp)[i];
    s  += v.x + v.y + v.z + v.w;
    ss += v.x * v.x + v.y * v.y + v.z * v.z + v.w * v.w;
  }
  __shared__ float rs[256], rq[256];
  rs[t] = s; rq[t] = ss;
  __syncthreads();
  for (int o = 128; o > 0; o >>= 1) {
    if (t < o) { rs[t] += rs[t + o]; rq[t] += rq[t + o]; }
    __syncthreads();
  }
  const float mean = rs[0] * (1.f / 16384.f);
  const float var  = rq[0] * (1.f / 16384.f) - mean * mean;
  const float rstd = rsqrtf(var + 1e-5f);

  _Float16* hp = h16 + ((size_t)b * CC + g * 16) * LL;
  for (int i = t; i < 4096; i += 256) {
    float4 v = ((const float4*)xp)[i];
    int e = i * 4;
    int c = g * 16 + (e >> 10);          // all 4 elements share the channel
    float sc = gw[c] * rstd;
    float sh = gb[c] - mean * sc;
    v4h o;
    o[0] = (_Float16)(v.x * sc + sh);
    o[1] = (_Float16)(v.y * sc + sh);
    o[2] = (_Float16)(v.z * sc + sh);
    o[3] = (_Float16)(v.w * sc + sh);
    ((v4h*)hp)[i] = o;
  }
}

// ===========================================================================
// Shared GEMM geometry for kernels 2 & 4:
//   block = 256 thr = 8 waves (4 m x 2 n); block tile 128(M) x 64(N);
//   wave tile 32x32 = 2x2 WMMA fragments; K-step 32.
//   A: f32 weights (M x K row-major) converted to f16 during LDS staging.
//   B: f16 activations (K x L row-major) staged transposed lBt[n][k].
// ===========================================================================

// Kernel 2: QKV GEMM. out[o,l] = sum_c qkv_w[o,c]*h16[c,l] + qkv_b[o];
// epilogue scatters q -> (bh,L,hd), kT -> (bh,hd,L), v -> (bh,L,hd) as f16.
__global__ __launch_bounds__(256)
void qkv_kernel(const float* __restrict__ wq, const float* __restrict__ bq,
                const _Float16* __restrict__ h16,
                _Float16* __restrict__ qbuf, _Float16* __restrict__ kTbuf,
                _Float16* __restrict__ vbuf) {
  const int nb = blockIdx.x * 64;
  const int mb = blockIdx.y * 128;
  const int b  = blockIdx.z;
  const int t    = threadIdx.x;
  const int lane = t & 31, wid = t >> 5;
  const int wm = wid >> 1, wn = wid & 1;       // 4 x 2 wave grid
  const int l15 = lane & 15, hi = lane >> 4;

  __shared__ alignas(32) _Float16 lA[128 * 32];    // [m][k]
  __shared__ alignas(32) _Float16 lBt[64 * 32];    // [n][k]
  const _Float16* hB = h16 + (size_t)b * CC * LL;

  v8f acc[2][2];
#pragma unroll
  for (int i = 0; i < 2; ++i)
#pragma unroll
    for (int j = 0; j < 2; ++j)
      acc[i][j] = (v8f){0.f, 0.f, 0.f, 0.f, 0.f, 0.f, 0.f, 0.f};

  for (int kb = 0; kb < CC; kb += 32) {
    // stage A: 128x32 f32 -> f16 LDS (1024 float4 chunks / 256 thr = 4 each)
#pragma unroll
    for (int it = 0; it < 4; ++it) {
      int idx = t + it * 256;
      int row = idx >> 3, ch = idx & 7;
      float4 v = *(const float4*)(wq + (size_t)(mb + row) * CC + kb + ch * 4);
      _Float16* dst = &lA[row * 32 + ch * 4];
      dst[0] = (_Float16)v.x; dst[1] = (_Float16)v.y;
      dst[2] = (_Float16)v.z; dst[3] = (_Float16)v.w;
    }
    // stage B transposed: 32(k) x 64(n) f16 -> lBt[n][k] (256 v8h chunks)
    {
      int row = t >> 3, n0 = (t & 7) * 8;          // row = k 0..31
      v8h v = *(const v8h*)(hB + (size_t)(kb + row) * LL + nb + n0);
#pragma unroll
      for (int e = 0; e < 8; ++e) lBt[(n0 + e) * 32 + row] = v[e];
    }
    __syncthreads();

    // fragments: 2 A (rows wm*32 + sm*16) and 2 B (cols wn*32 + sn*16)
    v16h af[2], bf[2];
#pragma unroll
    for (int sm = 0; sm < 2; ++sm) {
      const _Float16* arow = &lA[(wm * 32 + sm * 16 + l15) * 32];
      v8h a0 = *(const v8h*)(arow + hi * 16);
      v8h a1 = *(const v8h*)(arow + hi * 16 + 8);
#pragma unroll
      for (int i = 0; i < 8; ++i) { af[sm][i] = a0[i]; af[sm][i + 8] = a1[i]; }
    }
#pragma unroll
    for (int sn = 0; sn < 2; ++sn)
      bf[sn] = *(const v16h*)(&lBt[(wn * 32 + sn * 16 + l15) * 32 + hi * 16]);

#pragma unroll
    for (int sm = 0; sm < 2; ++sm)
#pragma unroll
      for (int sn = 0; sn < 2; ++sn)
        acc[sm][sn] = WMMA16x16x32(af[sm], bf[sn], acc[sm][sn]);
    __syncthreads();
  }

  // Epilogue: scatter to q / kT / v (f16) with bias
#pragma unroll
  for (int sm = 0; sm < 2; ++sm)
#pragma unroll
    for (int sn = 0; sn < 2; ++sn) {
      const int lidx = nb + wn * 32 + sn * 16 + l15;
#pragma unroll
      for (int r = 0; r < 8; ++r) {
        int o = mb + wm * 32 + sm * 16 + r + hi * 8;   // 0..1535
        float val = acc[sm][sn][r] + bq[o];
        int which = o >> 9;
        int c = o & (CC - 1);
        int head = c >> 6, d = c & (HD - 1);
        size_t bh = (size_t)(b * NH + head);
        if (which == 0)      qbuf[(bh * LL + lidx) * HD + d] = (_Float16)val;
        else if (which == 1) kTbuf[(bh * HD + d) * LL + lidx] = (_Float16)val;
        else                 vbuf[(bh * LL + lidx) * HD + d] = (_Float16)val;
      }
    }
}

// ===========================================================================
// Kernel 3: flash attention.  Block = 128 thr = 4 waves; 64 q-rows / block.
// Streams 32-key tiles: S = (Q*scale)K^T (WMMA), online softmax, O += P V.
// Output written transposed as hT (b, c=head*64+d, l) f16 for the proj GEMM.
// ===========================================================================
__global__ __launch_bounds__(128)
void attn_kernel(const _Float16* __restrict__ qbuf,
                 const _Float16* __restrict__ kTbuf,
                 const _Float16* __restrict__ vbuf,
                 _Float16* __restrict__ hT) {
  const int qt   = blockIdx.x;               // 0..15 (64-row q tiles)
  const int head = blockIdx.y;
  const int b    = blockIdx.z;
  const int t = threadIdx.x, lane = t & 31, w = t >> 5;
  const int l15 = lane & 15, hi = lane >> 4;
  const size_t bh = (size_t)(b * NH + head);

  __shared__ alignas(32) _Float16 lKt[32 * 64];     // [j][d]
  __shared__ alignas(32) _Float16 lVt[64 * 32];     // [d][j]
  __shared__ alignas(32) _Float16 lP[4][16 * 32];   // per-wave P bounce

  // Resident, pre-scaled Q fragments: 16 rows x 64 (two 16x32 A-frags)
  const int qrow = qt * 64 + w * 16 + l15;
  const _Float16* qp = qbuf + (bh * LL + qrow) * HD;
  const float scale = 0.125f;                 // 64^-0.5
  v16h qf[2];
#pragma unroll
  for (int j = 0; j < 2; ++j) {
    v8h c0 = *(const v8h*)(qp + j * 32 + hi * 16);
    v8h c1 = *(const v8h*)(qp + j * 32 + hi * 16 + 8);
#pragma unroll
    for (int i = 0; i < 8; ++i) {
      qf[j][i]     = (_Float16)((float)c0[i] * scale);
      qf[j][i + 8] = (_Float16)((float)c1[i] * scale);
    }
  }

  v8f oacc[4];
  float mrow[8], lrow[8];
#pragma unroll
  for (int f = 0; f < 4; ++f)
    oacc[f] = (v8f){0.f, 0.f, 0.f, 0.f, 0.f, 0.f, 0.f, 0.f};
#pragma unroll
  for (int r = 0; r < 8; ++r) { mrow[r] = -INFINITY; lrow[r] = 0.f; }

  for (int jb = 0; jb < LL; jb += 32) {
    // stage K tile -> lKt[j][d]  (from kT: (bh, d, l))
#pragma unroll
    for (int it = 0; it < 2; ++it) {
      int task = t + it * 128;                // 0..255
      int d = task >> 2, j0 = (task & 3) * 8;
      v8h v = *(const v8h*)(kTbuf + (bh * HD + d) * LL + jb + j0);
#pragma unroll
      for (int e = 0; e < 8; ++e) lKt[(j0 + e) * 64 + d] = v[e];
    }
    // stage V tile -> lVt[d][j]  (from v: (bh, l, d))
#pragma unroll
    for (int it = 0; it < 2; ++it) {
      int task = t + it * 128;
      int jr = task >> 3, d0 = (task & 7) * 8;
      v8h v = *(const v8h*)(vbuf + (bh * LL + jb + jr) * HD + d0);
#pragma unroll
      for (int e = 0; e < 8; ++e) lVt[(d0 + e) * 32 + jr] = v[e];
    }
    __syncthreads();

    // S tiles (16 x 32 keys): 2 n-subtiles x 2 k-steps
    v8f s0 = {0.f, 0.f, 0.f, 0.f, 0.f, 0.f, 0.f, 0.f};
    v8f s1 = {0.f, 0.f, 0.f, 0.f, 0.f, 0.f, 0.f, 0.f};
#pragma unroll
    for (int k = 0; k < 2; ++k) {
      v16h b0 = *(const v16h*)(&lKt[(l15) * 64 + k * 32 + hi * 16]);
      s0 = WMMA16x16x32(qf[k], b0, s0);
      v16h b1 = *(const v16h*)(&lKt[(16 + l15) * 64 + k * 32 + hi * 16]);
      s1 = WMMA16x16x32(qf[k], b1, s1);
    }

    // online softmax (rows live in 16-lane half-waves; xor 1/2/4/8 reduces)
    float p0[8], p1[8];
#pragma unroll
    for (int r = 0; r < 8; ++r) {
      float m = fmaxf(s0[r], s1[r]);
      m = fmaxf(m, __shfl_xor(m, 1, 32));
      m = fmaxf(m, __shfl_xor(m, 2, 32));
      m = fmaxf(m, __shfl_xor(m, 4, 32));
      m = fmaxf(m, __shfl_xor(m, 8, 32));
      float mn = fmaxf(mrow[r], m);
      float alpha = __expf(mrow[r] - mn);
      mrow[r] = mn;
      p0[r] = __expf(s0[r] - mn);
      p1[r] = __expf(s1[r] - mn);
      float rsum = p0[r] + p1[r];
      rsum += __shfl_xor(rsum, 1, 32);
      rsum += __shfl_xor(rsum, 2, 32);
      rsum += __shfl_xor(rsum, 4, 32);
      rsum += __shfl_xor(rsum, 8, 32);
      lrow[r] = lrow[r] * alpha + rsum;
#pragma unroll
      for (int f = 0; f < 4; ++f) oacc[f][r] *= alpha;
    }

    // P: C-layout f32 -> per-wave LDS -> A-fragment f16
    _Float16* pw = lP[w];
#pragma unroll
    for (int r = 0; r < 8; ++r) {
      int row = r + hi * 8;
      pw[row * 32 + l15]      = (_Float16)p0[r];
      pw[row * 32 + 16 + l15] = (_Float16)p1[r];
    }
    v8h c0 = *(const v8h*)(pw + l15 * 32 + hi * 16);
    v8h c1 = *(const v8h*)(pw + l15 * 32 + hi * 16 + 8);
    v16h pf;
#pragma unroll
    for (int i = 0; i < 8; ++i) { pf[i] = c0[i]; pf[i + 8] = c1[i]; }

    // O += P V : 4 d-subtiles, single k-step (K = 32 keys)
#pragma unroll
    for (int f = 0; f < 4; ++f) {
      v16h bv = *(const v16h*)(&lVt[(f * 16 + l15) * 32 + hi * 16]);
      oacc[f] = WMMA16x16x32(pf, bv, oacc[f]);
    }
    __syncthreads();
  }

  // normalize and write transposed: hT[(b*512 + head*64 + d)*1024 + l]
#pragma unroll
  for (int r = 0; r < 8; ++r) {
    float inv = 1.0f / lrow[r];
    int row = r + hi * 8;
    int lq = qt * 64 + w * 16 + row;
#pragma unroll
    for (int f = 0; f < 4; ++f) {
      int d = f * 16 + l15;
      hT[((size_t)b * CC + head * HD + d) * LL + lq] =
          (_Float16)(oacc[f][r] * inv);
    }
  }
}

// ===========================================================================
// Kernel 4: proj GEMM + bias + residual.  out = proj_w @ hT + proj_b + x
// Same 128x64 block tile / 2x2 wave blocking as the QKV GEMM.
// ===========================================================================
__global__ __launch_bounds__(256)
void proj_kernel(const float* __restrict__ wp, const float* __restrict__ bp,
                 const _Float16* __restrict__ hT, const float* __restrict__ x,
                 float* __restrict__ out) {
  const int nb = blockIdx.x * 64;
  const int mb = blockIdx.y * 128;
  const int b  = blockIdx.z;
  const int t    = threadIdx.x;
  const int lane = t & 31, wid = t >> 5;
  const int wm = wid >> 1, wn = wid & 1;
  const int l15 = lane & 15, hi = lane >> 4;

  __shared__ alignas(32) _Float16 lA[128 * 32];
  __shared__ alignas(32) _Float16 lBt[64 * 32];
  const _Float16* hB = hT + (size_t)b * CC * LL;

  v8f acc[2][2];
#pragma unroll
  for (int i = 0; i < 2; ++i)
#pragma unroll
    for (int j = 0; j < 2; ++j)
      acc[i][j] = (v8f){0.f, 0.f, 0.f, 0.f, 0.f, 0.f, 0.f, 0.f};

  for (int kb = 0; kb < CC; kb += 32) {
#pragma unroll
    for (int it = 0; it < 4; ++it) {
      int idx = t + it * 256;
      int row = idx >> 3, ch = idx & 7;
      float4 v = *(const float4*)(wp + (size_t)(mb + row) * CC + kb + ch * 4);
      _Float16* dst = &lA[row * 32 + ch * 4];
      dst[0] = (_Float16)v.x; dst[1] = (_Float16)v.y;
      dst[2] = (_Float16)v.z; dst[3] = (_Float16)v.w;
    }
    {
      int row = t >> 3, n0 = (t & 7) * 8;
      v8h v = *(const v8h*)(hB + (size_t)(kb + row) * LL + nb + n0);
#pragma unroll
      for (int e = 0; e < 8; ++e) lBt[(n0 + e) * 32 + row] = v[e];
    }
    __syncthreads();

    v16h af[2], bf[2];
#pragma unroll
    for (int sm = 0; sm < 2; ++sm) {
      const _Float16* arow = &lA[(wm * 32 + sm * 16 + l15) * 32];
      v8h a0 = *(const v8h*)(arow + hi * 16);
      v8h a1 = *(const v8h*)(arow + hi * 16 + 8);
#pragma unroll
      for (int i = 0; i < 8; ++i) { af[sm][i] = a0[i]; af[sm][i + 8] = a1[i]; }
    }
#pragma unroll
    for (int sn = 0; sn < 2; ++sn)
      bf[sn] = *(const v16h*)(&lBt[(wn * 32 + sn * 16 + l15) * 32 + hi * 16]);

#pragma unroll
    for (int sm = 0; sm < 2; ++sm)
#pragma unroll
      for (int sn = 0; sn < 2; ++sn)
        acc[sm][sn] = WMMA16x16x32(af[sm], bf[sn], acc[sm][sn]);
    __syncthreads();
  }

#pragma unroll
  for (int sm = 0; sm < 2; ++sm)
#pragma unroll
    for (int sn = 0; sn < 2; ++sn) {
      const int lidx = nb + wn * 32 + sn * 16 + l15;
#pragma unroll
      for (int r = 0; r < 8; ++r) {
        int o = mb + wm * 32 + sm * 16 + r + hi * 8;
        size_t idx = ((size_t)b * CC + o) * LL + lidx;
        out[idx] = acc[sm][sn][r] + bp[o] + x[idx];
      }
    }
}

// ===========================================================================
extern "C" void kernel_launch(void* const* d_in, const int* in_sizes, int n_in,
                              void* d_out, int out_size, void* d_ws, size_t ws_size,
                              hipStream_t stream) {
  const float* x      = (const float*)d_in[0];
  const float* norm_w = (const float*)d_in[1];
  const float* norm_b = (const float*)d_in[2];
  const float* qkv_w  = (const float*)d_in[3];
  const float* qkv_b  = (const float*)d_in[4];
  const float* proj_w = (const float*)d_in[5];
  const float* proj_b = (const float*)d_in[6];
  float* out = (float*)d_out;

  const size_t SEG = (size_t)16 * CC * LL;   // 8M f16 elements = 16 MB
  _Float16* h16 = (_Float16*)d_ws;
  _Float16* qb  = h16 + SEG;
  _Float16* kTb = qb  + SEG;
  _Float16* vb  = kTb + SEG;
  _Float16* hT  = vb  + SEG;

  gn_kernel  <<<dim3(32, 16),     256, 0, stream>>>(x, norm_w, norm_b, h16);
  qkv_kernel <<<dim3(16, 12, 16), 256, 0, stream>>>(qkv_w, qkv_b, h16, qb, kTb, vb);
  attn_kernel<<<dim3(16, 8, 16),  128, 0, stream>>>(qb, kTb, vb, hT);
  proj_kernel<<<dim3(16, 4, 16),  256, 0, stream>>>(proj_w, proj_b, hT, x, out);
}
